// SelfAttention_9861244911704
// MI455X (gfx1250) — matrix operand order repaired and checked by reading, not compiled, not run on hardware
//
#include <hip/hip_runtime.h>
#include <hip/hip_bf16.h>
#include <math.h>

#define N_HEADS 16
#define D_HEAD  64
#define D_MODEL 1024
#define SEQ_T   2048
#define BATCH   2

typedef __attribute__((ext_vector_type(16))) __bf16 v16bf;
typedef __attribute__((ext_vector_type(2)))  __bf16 v2bf;
typedef __attribute__((ext_vector_type(8)))  float  v8f;

union FragAB {
    v16bf bf;
    unsigned int u[8];
};
union FragC {
    v8f f;
    float x[8];
};

__device__ __forceinline__ unsigned short cvt_bf16(float f) {
    return __builtin_bit_cast(unsigned short, (__bf16)f);
}
__device__ __forceinline__ unsigned int pack2_bf16(float lo, float hi) {
    v2bf p;
    p.x = (__bf16)lo;
    p.y = (__bf16)hi;
    return __builtin_bit_cast(unsigned int, p);
}
__device__ __forceinline__ v8f wmma_bf16(const FragAB& a, const FragAB& b, v8f c) {
    // D(f32 16x16) = A(bf16 16x32) * B(bf16 32x16) + C
    return __builtin_amdgcn_wmma_f32_16x16x32_bf16(false, a.bf, false, b.bf,
                                                   (short)0, c, false, false);
}

// CDNA5 async global->LDS copy (16B per lane), tracked by ASYNCcnt.
// Low 32 bits of a generic pointer to __shared__ are the LDS-relative address
// (ISA 10.2).
__device__ __forceinline__ void async_b128(void* lds_ptr, const void* gptr) {
    unsigned int       l = (unsigned int)(size_t)lds_ptr;
    unsigned long long g = (unsigned long long)(size_t)gptr;
    asm volatile("global_load_async_to_lds_b128 %0, %1, off"
                 :: "v"(l), "v"(g) : "memory");
}
__device__ __forceinline__ void wait_async() {
    asm volatile("s_wait_asynccnt 0" ::: "memory");
}

// ---------------------------------------------------------------------------
// Kernel 1: fused QKV projection + RoPE.
//   q,k -> [B,H,T,D] bf16 (RoPE applied);  v -> [B,H,D,T] bf16 (transposed)
// grid(16, 64, 3), block 256
// LDS tiles hold bf16 pairs already in WMMA register layout.
// ---------------------------------------------------------------------------
__global__ void __launch_bounds__(256)
qkv_rope_kernel(const float* __restrict__ x,
                const float* __restrict__ Wq,
                const float* __restrict__ Wk,
                const float* __restrict__ Wv,
                const float* __restrict__ cosb,
                const float* __restrict__ sinb,
                unsigned short* __restrict__ qo,
                unsigned short* __restrict__ ko,
                unsigned short* __restrict__ vo) {
    __shared__ unsigned int Ap[64][20];  // [m][pair(k)]  bf16 pairs
    __shared__ unsigned int Bp[64][20];  // [n][pair(k)]  bf16 pairs
    __shared__ float Ct[64][66];         // output tile f32

    const int n0  = blockIdx.x * 64;
    const int m0  = blockIdx.y * 64;
    const int mat = blockIdx.z;
    const float* W        = (mat == 0) ? Wq : (mat == 1) ? Wk : Wv;
    unsigned short* outp  = (mat == 0) ? qo : (mat == 1) ? ko : vo;

    const int tid  = threadIdx.x;
    const int wave = tid >> 5;
    const int lane = tid & 31;
    const int li   = lane & 15;
    const int g    = lane >> 4;
    const int iblk = wave >> 1;          // 0..3 : M sub-block
    const int j0   = (wave & 1) * 2;     // 0 or 2 : N sub-block base

    FragC acc[2];
#pragma unroll
    for (int jj = 0; jj < 2; ++jj)
#pragma unroll
        for (int r = 0; r < 8; ++r) acc[jj].x[r] = 0.0f;

    const int arow = tid >> 2;           // 0..63
    const int aseg = (tid & 3) * 8;      // 0,8,16,24 (float offset)
    const int bkp  = tid >> 4;           // 0..15 : k-pair row
    const int bn4  = (tid & 15) * 4;     // 0..60 : n base

    for (int k0 = 0; k0 < D_MODEL; k0 += 32) {
        __syncthreads();
        {   // A tile: x[m0..+63][k0..+31]  -> bf16 pairs
            const float* src = x + (size_t)(m0 + arow) * D_MODEL + k0 + aseg;
            float4 a0 = *(const float4*)(src);
            float4 a1 = *(const float4*)(src + 4);
            unsigned int p0 = pack2_bf16(a0.x, a0.y);
            unsigned int p1 = pack2_bf16(a0.z, a0.w);
            unsigned int p2 = pack2_bf16(a1.x, a1.y);
            unsigned int p3 = pack2_bf16(a1.z, a1.w);
            *(uint4*)&Ap[arow][(tid & 3) * 4] = make_uint4(p0, p1, p2, p3);
        }
        {   // B tile: W[k0..+31][n0..+63] -> Bp[n][k/2] with (k,k+1) packed
            const float* s0 = W + (size_t)(k0 + 2 * bkp) * D_MODEL + n0 + bn4;
            const float* s1 = s0 + D_MODEL;
            float4 r0 = *(const float4*)s0;
            float4 r1 = *(const float4*)s1;
            Bp[bn4 + 0][bkp] = pack2_bf16(r0.x, r1.x);
            Bp[bn4 + 1][bkp] = pack2_bf16(r0.y, r1.y);
            Bp[bn4 + 2][bkp] = pack2_bf16(r0.z, r1.z);
            Bp[bn4 + 3][bkp] = pack2_bf16(r0.w, r1.w);
        }
        if (k0 + 32 < D_MODEL) {   // prefetch next chunk into near cache
            __builtin_prefetch(x + (size_t)(m0 + arow) * D_MODEL + k0 + 32 + aseg, 0, 3);
            __builtin_prefetch(W + (size_t)(k0 + 32 + 2 * bkp) * D_MODEL + n0 + bn4, 0, 3);
        }
        __syncthreads();

        FragAB a;
        const int am = iblk * 16 + li;
#pragma unroll
        for (int v = 0; v < 8; ++v)       // ISA A 16x32: pair = (v>>2)*8+g*4+(v&3)
            a.u[v] = Ap[am][((v >> 2) * 8) + g * 4 + (v & 3)];
#pragma unroll
        for (int jj = 0; jj < 2; ++jj) {
            FragAB b;
            const int bnn = (j0 + jj) * 16 + li;
#pragma unroll
            for (int v = 0; v < 8; ++v)   // ISA B 32x16: pair = (v>>2)*8+g*4+(v&3)
                b.u[v] = Bp[bnn][((v >> 2) * 8) + g * 4 + (v & 3)];
            acc[jj].f = wmma_bf16(a, b, acc[jj].f);
        }
    }

    __syncthreads();
#pragma unroll
    for (int jj = 0; jj < 2; ++jj)
#pragma unroll
        for (int r = 0; r < 8; ++r)
            Ct[iblk * 16 + r + g * 8][(j0 + jj) * 16 + li] = acc[jj].x[r];
    __syncthreads();

    // RoPE (q,k only) + store. N tile of 64 == one full head.
    {
        const int rr   = tid >> 2;
        const int dsg  = (tid & 3) * 16;
        const int mrow = m0 + rr;
        const int bidx = mrow >> 11;            // / SEQ_T
        const int ttok = mrow & (SEQ_T - 1);
        const int h    = n0 >> 6;

        float vals[16];
#pragma unroll
        for (int q = 0; q < 16; ++q) {
            int d = dsg + q;
            float vcur = Ct[rr][d];
            if (mat < 2) {
                int dr = d & 31;
                float cv = cosb[ttok * 32 + dr];
                float sv = sinb[ttok * 32 + dr];
                float other = Ct[rr][(d < 32) ? d + 32 : d - 32];
                vals[q] = (d < 32) ? (vcur * cv - other * sv)
                                   : (other * sv + vcur * cv);
            } else {
                vals[q] = vcur;
            }
        }
        if (mat < 2) {
            size_t base = (((size_t)(bidx * N_HEADS + h) * SEQ_T) + ttok) * D_HEAD + dsg;
            unsigned int pk[8];
#pragma unroll
            for (int p = 0; p < 8; ++p) pk[p] = pack2_bf16(vals[2*p], vals[2*p+1]);
            uint4* dst = (uint4*)(outp + base);
            dst[0] = make_uint4(pk[0], pk[1], pk[2], pk[3]);
            dst[1] = make_uint4(pk[4], pk[5], pk[6], pk[7]);
        } else {
            // V stored transposed: [B,H,D,T]
#pragma unroll
            for (int q = 0; q < 16; ++q) {
                int d = dsg + q;
                size_t idx = (((size_t)(bidx * N_HEADS + h) * D_HEAD) + d) * SEQ_T + ttok;
                outp[idx] = cvt_bf16(vals[q]);
            }
        }
    }
}

// ---------------------------------------------------------------------------
// Kernel 2: causal flash attention per (b,h, 64-row q tile).
// grid(32, 32), block 256.  Q/K/V tiles use async global->LDS DMA.
// ---------------------------------------------------------------------------
__global__ void __launch_bounds__(256)
attn_kernel(const unsigned short* __restrict__ qg,
            const unsigned short* __restrict__ kg,
            const unsigned short* __restrict__ vtg,
            unsigned short* __restrict__ zo) {
    __shared__ unsigned int Qs[64][36];   // [q][pair(d)]     bf16 pairs
    __shared__ unsigned int Ks[64][36];   // [key][pair(d)]
    __shared__ unsigned int Vs[64][36];   // [d][pair(key)]
    __shared__ unsigned int Ps[64][36];   // [q][pair(key)]
    __shared__ float Ss[64][66];          // scores f32
    __shared__ float alphaS[64];
    __shared__ float linvS[64];

    const int qi = blockIdx.x;
    const int bh = blockIdx.y;
    const int q0 = qi * 64;

    const unsigned short* Qbase = qg  + (size_t)bh * SEQ_T * D_HEAD;
    const unsigned short* Kbase = kg  + (size_t)bh * SEQ_T * D_HEAD;
    const unsigned short* Vbase = vtg + (size_t)bh * D_HEAD * SEQ_T;  // [d][t]

    const int tid  = threadIdx.x;
    const int wave = tid >> 5;
    const int lane = tid & 31;
    const int li   = lane & 15;
    const int g    = lane >> 4;
    const int iblk = wave >> 1;
    const int j0   = (wave & 1) * 2;

    const int sr   = tid >> 2;   // softmax row 0..63
    const int sseg = tid & 3;    // 16-wide column segment

    {   // async load Q tile (bf16 pairs, d-contiguous): 32B per lane
        const unsigned short* src = Qbase + (size_t)(q0 + sr) * D_HEAD + sseg * 16;
        async_b128(&Qs[sr][sseg * 8],     src);
        async_b128(&Qs[sr][sseg * 8 + 4], src + 8);
    }
    wait_async();
    __syncthreads();

    FragAB qa[2];
#pragma unroll
    for (int ck = 0; ck < 2; ++ck)
#pragma unroll
        for (int v = 0; v < 8; ++v) {
            int pr = ck * 16 + ((v >> 2) * 8) + g * 4 + (v & 3);
            qa[ck].u[v] = Qs[iblk * 16 + li][pr];
        }

    FragC oc[2];
#pragma unroll
    for (int jj = 0; jj < 2; ++jj)
#pragma unroll
        for (int r = 0; r < 8; ++r) oc[jj].x[r] = 0.0f;

    float m_run = -3.0e38f, l_run = 0.0f;
    const int qglob = q0 + sr;
    const float scale = 0.125f;   // 1/sqrt(64)

    for (int kt = 0; kt <= qi; ++kt) {
        __syncthreads();
        {   // async load K tile + V tile (pure bf16 copies, no VGPR round-trip)
            const unsigned short* srcK = Kbase + (size_t)(kt * 64 + sr) * D_HEAD + sseg * 16;
            async_b128(&Ks[sr][sseg * 8],     srcK);
            async_b128(&Ks[sr][sseg * 8 + 4], srcK + 8);
            const unsigned short* srcV = Vbase + (size_t)sr * SEQ_T + kt * 64 + sseg * 16;
            async_b128(&Vs[sr][sseg * 8],     srcV);
            async_b128(&Vs[sr][sseg * 8 + 4], srcV + 8);
        }
        wait_async();
        __syncthreads();

        // S = Q * K^T
        FragC sc[2];
#pragma unroll
        for (int jj = 0; jj < 2; ++jj)
#pragma unroll
            for (int r = 0; r < 8; ++r) sc[jj].x[r] = 0.0f;
#pragma unroll
        for (int ck = 0; ck < 2; ++ck) {
#pragma unroll
            for (int jj = 0; jj < 2; ++jj) {
                FragAB kb;
#pragma unroll
                for (int v = 0; v < 8; ++v)
                    kb.u[v] = Ks[(j0 + jj) * 16 + li][ck * 16 + ((v >> 2) * 8) + g * 4 + (v & 3)];
                sc[jj].f = wmma_bf16(qa[ck], kb, sc[jj].f);
            }
        }
#pragma unroll
        for (int jj = 0; jj < 2; ++jj)
#pragma unroll
            for (int r = 0; r < 8; ++r)
                Ss[iblk * 16 + r + g * 8][(j0 + jj) * 16 + li] = sc[jj].x[r] * scale;
        __syncthreads();

        // online softmax (4 lanes per row, shuffle reduce)
        {
            const int kb0 = kt * 64 + sseg * 16;
            float sv[16];
            float vmax = -3.0e38f;
#pragma unroll
            for (int t = 0; t < 16; ++t) {
                float s = Ss[sr][sseg * 16 + t];
                sv[t] = (kb0 + t <= qglob) ? s : -3.0e38f;
                vmax = fmaxf(vmax, sv[t]);
            }
            vmax = fmaxf(vmax, __shfl_xor(vmax, 1, 32));
            vmax = fmaxf(vmax, __shfl_xor(vmax, 2, 32));
            float mnew = fmaxf(m_run, vmax);
            float psum = 0.0f;
            unsigned int pk[8];
#pragma unroll
            for (int p = 0; p < 8; ++p) {
                float e0 = (kb0 + 2*p     <= qglob) ? __expf(sv[2*p]   - mnew) : 0.0f;
                float e1 = (kb0 + 2*p + 1 <= qglob) ? __expf(sv[2*p+1] - mnew) : 0.0f;
                psum += e0 + e1;
                pk[p] = pack2_bf16(e0, e1);
            }
            psum += __shfl_xor(psum, 1, 32);
            psum += __shfl_xor(psum, 2, 32);
            float alpha = (mnew > m_run) ? __expf(m_run - mnew) : 1.0f;
            l_run = l_run * alpha + psum;
            m_run = mnew;
            if (sseg == 0) alphaS[sr] = alpha;
            unsigned int* d = &Ps[sr][sseg * 8];
#pragma unroll
            for (int p = 0; p < 8; ++p) d[p] = pk[p];
        }
        __syncthreads();

        // O = O*alpha + P @ V
#pragma unroll
        for (int jj = 0; jj < 2; ++jj)
#pragma unroll
            for (int r = 0; r < 8; ++r)
                oc[jj].x[r] *= alphaS[iblk * 16 + r + g * 8];
#pragma unroll
        for (int ck = 0; ck < 2; ++ck) {
            FragAB pa;
#pragma unroll
            for (int v = 0; v < 8; ++v) {
                int pr = ck * 16 + ((v >> 2) * 8) + g * 4 + (v & 3);
                pa.u[v] = Ps[iblk * 16 + li][pr];
            }
#pragma unroll
            for (int jj = 0; jj < 2; ++jj) {
                FragAB vb;
#pragma unroll
                for (int v = 0; v < 8; ++v)
                    vb.u[v] = Vs[(j0 + jj) * 16 + li][ck * 16 + ((v >> 2) * 8) + g * 4 + (v & 3)];
                oc[jj].f = wmma_bf16(pa, vb, oc[jj].f);
            }
        }
    }

    if (sseg == 0) linvS[sr] = 1.0f / l_run;
    __syncthreads();

    // write O -> z [B, T, H*D] bf16
    const int b = bh >> 4;
    const int h = bh & 15;
#pragma unroll
    for (int jj = 0; jj < 2; ++jj)
#pragma unroll
        for (int r = 0; r < 8; ++r) {
            int row = iblk * 16 + r + g * 8;
            float val = oc[jj].x[r] * linvS[row];
            size_t idx = ((size_t)(b * SEQ_T) + q0 + row) * (N_HEADS * D_HEAD)
                       + h * D_HEAD + (j0 + jj) * 16 + li;
            zo[idx] = cvt_bf16(val);
        }
}

// ---------------------------------------------------------------------------
// Kernel 3: out projection  out(f32) = z(bf16) @ W_out
// grid(16, 64), block 256.  A tile uses async global->LDS DMA.
// ---------------------------------------------------------------------------
__global__ void __launch_bounds__(256)
outproj_kernel(const unsigned short* __restrict__ z,
               const float* __restrict__ Wout,
               float* __restrict__ out) {
    __shared__ unsigned int Az[64][20];   // [m][pair(k)] bf16 pairs
    __shared__ unsigned int Bp[64][20];   // [n][pair(k)] bf16 pairs
    __shared__ float Ct[64][66];

    const int n0 = blockIdx.x * 64;
    const int m0 = blockIdx.y * 64;

    const int tid  = threadIdx.x;
    const int wave = tid >> 5;
    const int lane = tid & 31;
    const int li   = lane & 15;
    const int g    = lane >> 4;
    const int iblk = wave >> 1;
    const int j0   = (wave & 1) * 2;

    FragC acc[2];
#pragma unroll
    for (int jj = 0; jj < 2; ++jj)
#pragma unroll
        for (int r = 0; r < 8; ++r) acc[jj].x[r] = 0.0f;

    const int ar  = tid >> 2;          // 0..63
    const int as  = tid & 3;           // 0..3
    const int bkp = tid >> 4;          // 0..15
    const int bn4 = (tid & 15) * 4;    // 0..60

    for (int k0 = 0; k0 < D_MODEL; k0 += 32) {
        __syncthreads();
        // A tile (already bf16): async copy z[m0..+63][k0..+31] -> LDS
        async_b128(&Az[ar][as * 4], z + (size_t)(m0 + ar) * D_MODEL + k0 + as * 8);
        {   // B tile: Wout[k0..+31][n0..+63] -> pairs (f32 -> bf16 convert)
            const float* s0 = Wout + (size_t)(k0 + 2 * bkp) * D_MODEL + n0 + bn4;
            const float* s1 = s0 + D_MODEL;
            float4 r0 = *(const float4*)s0;
            float4 r1 = *(const float4*)s1;
            Bp[bn4 + 0][bkp] = pack2_bf16(r0.x, r1.x);
            Bp[bn4 + 1][bkp] = pack2_bf16(r0.y, r1.y);
            Bp[bn4 + 2][bkp] = pack2_bf16(r0.z, r1.z);
            Bp[bn4 + 3][bkp] = pack2_bf16(r0.w, r1.w);
        }
        if (k0 + 32 < D_MODEL) {
            __builtin_prefetch(z + (size_t)(m0 + ar) * D_MODEL + k0 + 32, 0, 3);
            __builtin_prefetch(Wout + (size_t)(k0 + 32 + 2 * bkp) * D_MODEL + n0 + bn4, 0, 3);
        }
        wait_async();
        __syncthreads();

        FragAB a;
#pragma unroll
        for (int v = 0; v < 8; ++v)
            a.u[v] = Az[iblk * 16 + li][((v >> 2) * 8) + g * 4 + (v & 3)];
#pragma unroll
        for (int jj = 0; jj < 2; ++jj) {
            FragAB b;
            const int bnn = (j0 + jj) * 16 + li;
#pragma unroll
            for (int v = 0; v < 8; ++v)
                b.u[v] = Bp[bnn][((v >> 2) * 8) + g * 4 + (v & 3)];
            acc[jj].f = wmma_bf16(a, b, acc[jj].f);
        }
    }

    __syncthreads();
#pragma unroll
    for (int jj = 0; jj < 2; ++jj)
#pragma unroll
        for (int r = 0; r < 8; ++r)
            Ct[iblk * 16 + r + g * 8][(j0 + jj) * 16 + li] = acc[jj].x[r];
    __syncthreads();

    {   // coalesced f32 store
        const int rr  = tid >> 2;
        const int dsg = (tid & 3) * 16;
        float* dst = out + (size_t)(m0 + rr) * D_MODEL + n0 + dsg;
#pragma unroll
        for (int p = 0; p < 4; ++p) {
            float4 v4;
            v4.x = Ct[rr][dsg + p*4 + 0];
            v4.y = Ct[rr][dsg + p*4 + 1];
            v4.z = Ct[rr][dsg + p*4 + 2];
            v4.w = Ct[rr][dsg + p*4 + 3];
            *(float4*)(dst + p*4) = v4;
        }
    }
}

// ---------------------------------------------------------------------------
extern "C" void kernel_launch(void* const* d_in, const int* in_sizes, int n_in,
                              void* d_out, int out_size, void* d_ws, size_t ws_size,
                              hipStream_t stream) {
    (void)in_sizes; (void)n_in; (void)out_size; (void)ws_size;

    const float* x    = (const float*)d_in[0];
    const float* cosb = (const float*)d_in[1];
    const float* sinb = (const float*)d_in[2];
    const float* Wq   = (const float*)d_in[3];
    const float* Wk   = (const float*)d_in[4];
    const float* Wv   = (const float*)d_in[5];
    const float* Wout = (const float*)d_in[6];
    float* out = (float*)d_out;

    const size_t elems = (size_t)BATCH * N_HEADS * SEQ_T * D_HEAD;  // 4M
    unsigned short* qw  = (unsigned short*)d_ws;    // ws usage: 4 * 8MB = 32MB
    unsigned short* kw  = qw + elems;
    unsigned short* vtw = kw + elems;
    unsigned short* zw  = vtw + elems;

    qkv_rope_kernel<<<dim3(D_MODEL / 64, (BATCH * SEQ_T) / 64, 3), 256, 0, stream>>>(
        x, Wq, Wk, Wv, cosb, sinb, qw, kw, vtw);
    attn_kernel<<<dim3(SEQ_T / 64, BATCH * N_HEADS), 256, 0, stream>>>(qw, kw, vtw, zw);
    outproj_kernel<<<dim3(D_MODEL / 64, (BATCH * SEQ_T) / 64), 256, 0, stream>>>(
        zw, Wout, out);
}
